// LinkPredictionGNN_7241314861683
// MI455X (gfx1250) — matrix operand order repaired and checked by reading, not compiled, not run on hardware
//
#include <hip/hip_runtime.h>
#include <hip/hip_bf16.h>

#define NNODES 100000
#define NEDGES 800000
#define CH     128
#define OUTC   64
#define GN_EPS 1e-5f

typedef __attribute__((ext_vector_type(16))) __bf16 v16bf;
typedef __attribute__((ext_vector_type(8)))  __bf16 v8bf;
typedef __attribute__((ext_vector_type(4)))  __bf16 v4bf;
typedef __attribute__((ext_vector_type(8)))  float  v8f;

__device__ __forceinline__ __bf16 f2bf(float f) { return (__bf16)f; }

// ---------------- degree / dinv ----------------
__global__ void deg_init_kernel(float* __restrict__ deg, int n) {
    int i = blockIdx.x * blockDim.x + threadIdx.x;
    if (i < n) deg[i] = 1.0f;   // self-loop
}

__global__ void deg_accum_kernel(const long long* __restrict__ ei, float* __restrict__ deg, int e) {
    int i = blockIdx.x * blockDim.x + threadIdx.x;
    if (i < e) atomicAdd(&deg[(int)ei[(size_t)e + i]], 1.0f);  // dst row of edge_index
}

__global__ void dinv_kernel(float* __restrict__ deg, int n) {
    int i = blockIdx.x * blockDim.x + threadIdx.x;
    if (i < n) deg[i] = rsqrtf(deg[i]);   // deg >= 1 always
}

// ---------------- pack W (fp32 [128,NC]) into bf16 WMMA B-fragment order ----------------
// Wp[((wave*4 + kk)*32 + lane)*16 + i] = bf16( W[kk*32 + (lane>>4)*8 + (i&7) + (i>>3)*16][wave*16 + (lane&15)] )
template<int NC>
__global__ __launch_bounds__(256)
void pack_w_kernel(const float* __restrict__ W, __bf16* __restrict__ Wp) {
    int idx = blockIdx.x * blockDim.x + threadIdx.x;
    if (idx >= CH * NC) return;
    int i  = idx & 15;
    int l  = (idx >> 4) & 31;
    int t  = idx >> 9;          // wave*4 + kk
    int kk = t & 3;
    int w  = t >> 2;
    int k  = kk * 32 + ((l >> 4) << 3) + (i & 7) + ((i >> 3) << 4);
    int n  = w * 16 + (l & 15);
    Wp[idx] = f2bf(W[(size_t)k * NC + n]);
}

// ---------------- WMMA GEMM: [N,128] x [128,NC] ----------------
// FINAL==false: out0 = out1 = dinv[row] * (A@W)   (pre-scaled messages + self-loop init)
// FINAL==true : out0 = (A@W) + bias[col]
template<int NC, bool FINAL>
__global__ __launch_bounds__(NC * 2)
void gemm_wmma_kernel(const float* __restrict__ A, const __bf16* __restrict__ Wp,
                      const float* __restrict__ sb,   // dinv[N] or bias[NC]
                      float* __restrict__ out0, float* __restrict__ out1)
{
    __shared__ __align__(16) __bf16 As[16 * CH];   // 16 rows x K=128, bf16 (4 KB)
    const int tid  = threadIdx.x;
    const int wave = tid >> 5;
    const int lane = tid & 31;
    const int row0 = blockIdx.x * 16;

    // cooperative fp32 -> bf16 load of the 16x128 A tile (contiguous 2048-float span)
    const float4* Ag = (const float4*)(A + (size_t)row0 * CH);
    for (int i = tid; i < (16 * CH) / 4; i += NC * 2) {
        float4 v = Ag[i];
        v4bf t;
        t[0] = f2bf(v.x); t[1] = f2bf(v.y); t[2] = f2bf(v.z); t[3] = f2bf(v.w);
        *(v4bf*)&As[i * 4] = t;
    }
    __syncthreads();

    const int m    = lane & 15;
    const int kgrp = lane >> 4;             // 0: lanes 0-15, 1: lanes 16-31
    const int ncol = wave * 16 + (lane & 15);
    const v16bf* Wp16 = (const v16bf*)Wp;   // 32-byte fragments
    v8f acc = {};

    #pragma unroll
    for (int kk = 0; kk < CH / 32; ++kk) {
        const int k0 = kk * 32;
        // A frag: two contiguous 8-element (16 B) LDS chunks per lane
        const int abase = m * CH + k0 + (kgrp << 3);
        v8bf lo = *(const v8bf*)&As[abase];
        v8bf hi = *(const v8bf*)&As[abase + 16];
        v16bf a = __builtin_shufflevector(lo, hi,
                    0, 1, 2, 3, 4, 5, 6, 7, 8, 9, 10, 11, 12, 13, 14, 15);
        // B frag: one 32-byte contiguous load from pre-packed weights
        v16bf b = Wp16[(wave * 4 + kk) * 32 + lane];
        acc = __builtin_amdgcn_wmma_f32_16x16x32_bf16(
            false, a, false, b, (short)0, acc, false, false);
    }

    // epilogue: C/D layout — VGPR r holds row (kgrp*8 + r), lane gives col
    #pragma unroll
    for (int r = 0; r < 8; ++r) {
        int grow = row0 + (kgrp << 3) + r;
        float v  = acc[r];
        if (FINAL) {
            out0[(size_t)grow * NC + ncol] = v + sb[ncol];
        } else {
            v *= sb[grow];
            out0[(size_t)grow * NC + ncol] = v;
            out1[(size_t)grow * NC + ncol] = v;   // self-loop contribution
        }
    }
}

// ---------------- edge scatter: accum[dst] += G[src], one wave per edge ----------------
__global__ __launch_bounds__(256)
void scatter_kernel(const float4* __restrict__ G, float* __restrict__ acc,
                    const long long* __restrict__ ei, int e)
{
    int gid  = blockIdx.x * blockDim.x + threadIdx.x;
    int edge = gid >> 5;
    int lane = threadIdx.x & 31;
    if (edge >= e) return;
    int s = (int)ei[edge];
    int d = (int)ei[(size_t)e + edge];
    float4 v = G[(size_t)s * (CH / 4) + lane];
    float* p = acc + (size_t)d * CH + lane * 4;
    atomicAdd(p + 0, v.x);
    atomicAdd(p + 1, v.y);
    atomicAdd(p + 2, v.z);
    atomicAdd(p + 3, v.w);
}

// ---------------- zero the stats buffer ----------------
__global__ void zero_stats_kernel(float* __restrict__ stats) {
    stats[threadIdx.x] = 0.0f;
}

// ---------------- in-place finalize (dinv scale + bias) and column stats ----------------
__global__ __launch_bounds__(CH)
void finalize_stats_kernel(float* __restrict__ H, const float* __restrict__ dinv,
                           const float* __restrict__ bias, float* __restrict__ stats, int n)
{
    int c = threadIdx.x;
    float bc = bias[c];
    float s = 0.f, s2 = 0.f;
    for (int r = blockIdx.x; r < n; r += gridDim.x) {
        float v = dinv[r] * H[(size_t)r * CH + c] + bc;
        H[(size_t)r * CH + c] = v;
        s  += v;
        s2 += v * v;
    }
    atomicAdd(&stats[c],      s);
    atomicAdd(&stats[CH + c], s2);
}

// ---------------- GraphNorm + ReLU (+ optional residual add) ----------------
template<bool RES>
__global__ __launch_bounds__(256)
void norm_relu_kernel(const float* __restrict__ H, const float* __restrict__ stats,
                      const float* __restrict__ w, const float* __restrict__ b,
                      const float* __restrict__ ms, const float* __restrict__ res,
                      float* __restrict__ out, int total)
{
    int i = blockIdx.x * blockDim.x + threadIdx.x;
    if (i >= total) return;
    int c = i & (CH - 1);
    const float invN = 1.0f / (float)NNODES;
    float mean = stats[c] * invN;
    float e2   = stats[CH + c] * invN;
    float m    = ms[c];
    // E[(h - ms*mean)^2] = E[h^2] - 2*ms*mean*E[h] + ms^2*mean^2
    float var = e2 - 2.0f * m * mean * mean + m * m * mean * mean;
    float rs  = rsqrtf(var + GN_EPS);
    float v   = w[c] * (H[i] - m * mean) * rs + b[c];
    v = fmaxf(v, 0.0f);
    if (RES) v += res[i];
    out[i] = v;
}

extern "C" void kernel_launch(void* const* d_in, const int* in_sizes, int n_in,
                              void* d_out, int out_size, void* d_ws, size_t ws_size,
                              hipStream_t stream) {
    (void)in_sizes; (void)n_in; (void)out_size; (void)ws_size;

    const float*     x      = (const float*)d_in[0];
    const long long* ei     = (const long long*)d_in[1];
    const float*     W1     = (const float*)d_in[2];
    const float*     b1     = (const float*)d_in[3];
    const float*     W2     = (const float*)d_in[4];
    const float*     b2     = (const float*)d_in[5];
    const float*     gn1_w  = (const float*)d_in[6];
    const float*     gn1_b  = (const float*)d_in[7];
    const float*     gn1_ms = (const float*)d_in[8];
    const float*     gn2_w  = (const float*)d_in[9];
    const float*     gn2_b  = (const float*)d_in[10];
    const float*     gn2_ms = (const float*)d_in[11];
    const float*     Wr     = (const float*)d_in[12];
    const float*     br     = (const float*)d_in[13];
    float*           out    = (float*)d_out;

    // workspace layout (floats, all region starts 32B-aligned)
    float* ws    = (float*)d_ws;
    float* dinv  = ws;                                   // N
    float* stats = dinv + NNODES;                        // 256
    float* G     = stats + 256;                          // N*128  (messages / xsum)
    float* accum = G + (size_t)NNODES * CH;              // N*128
    float* x1    = accum + (size_t)NNODES * CH;          // N*128
    __bf16* Wp   = (__bf16*)(x1 + (size_t)NNODES * CH);  // 128*128 bf16 packed weights

    const size_t NC128 = (size_t)NNODES * CH;
    const int nb256 = (NNODES + 255) / 256;
    const int eb256 = (NEDGES + 255) / 256;

    // degrees (with self loops) -> dinv
    deg_init_kernel<<<nb256, 256, 0, stream>>>(dinv, NNODES);
    deg_accum_kernel<<<eb256, 256, 0, stream>>>(ei, dinv, NEDGES);
    dinv_kernel<<<nb256, 256, 0, stream>>>(dinv, NNODES);

    // ---- layer 1 ----
    pack_w_kernel<CH><<<(CH * CH) / 256, 256, 0, stream>>>(W1, Wp);
    gemm_wmma_kernel<CH, false><<<NNODES / 16, 2 * CH, 0, stream>>>(x, Wp, dinv, G, accum);
    scatter_kernel<<<NEDGES / 8, 256, 0, stream>>>((const float4*)G, accum, ei, NEDGES);
    zero_stats_kernel<<<1, 256, 0, stream>>>(stats);
    finalize_stats_kernel<<<512, CH, 0, stream>>>(accum, dinv, b1, stats, NNODES);
    norm_relu_kernel<false><<<(int)(NC128 / 256), 256, 0, stream>>>(
        accum, stats, gn1_w, gn1_b, gn1_ms, nullptr, x1, (int)NC128);

    // ---- layer 2 ----
    pack_w_kernel<CH><<<(CH * CH) / 256, 256, 0, stream>>>(W2, Wp);
    gemm_wmma_kernel<CH, false><<<NNODES / 16, 2 * CH, 0, stream>>>(x1, Wp, dinv, G, accum);
    scatter_kernel<<<NEDGES / 8, 256, 0, stream>>>((const float4*)G, accum, ei, NEDGES);
    zero_stats_kernel<<<1, 256, 0, stream>>>(stats);
    finalize_stats_kernel<<<512, CH, 0, stream>>>(accum, dinv, b2, stats, NNODES);
    // G := x1 + relu(norm(...))   (residual-fused, G is free after scatter)
    norm_relu_kernel<true><<<(int)(NC128 / 256), 256, 0, stream>>>(
        accum, stats, gn2_w, gn2_b, gn2_ms, x1, G, (int)NC128);

    // ---- final projection: out = G @ Wr + br ----
    pack_w_kernel<OUTC><<<(CH * OUTC) / 256, 256, 0, stream>>>(Wr, Wp);
    gemm_wmma_kernel<OUTC, true><<<NNODES / 16, 2 * OUTC, 0, stream>>>(G, Wp, br, out, nullptr);
}
